// MultiheadAttention_15058155339957
// MI455X (gfx1250) — matrix operand order repaired and checked by reading, not compiled, not run on hardware
//
#include <hip/hip_runtime.h>
#include <hip/hip_bf16.h>
#include <math.h>

#define T_DIM 512
#define B_DIM 8
#define E_DIM 1024
#define H_DIM 32
#define DH    32
#define NEXP_ 256
#define S_DIM 768
#define SCALING 0.17677669529663687f
#define LN_EPS_ 1e-5f
#define NEG_BIG -3.0e38f

typedef __bf16 bf16_t;
typedef bf16_t v16bf __attribute__((ext_vector_type(16)));
typedef float  v8f   __attribute__((ext_vector_type(8)));
typedef unsigned int v4u __attribute__((ext_vector_type(4)));
typedef int v8i __attribute__((ext_vector_type(8)));
typedef int v4i __attribute__((ext_vector_type(4)));

struct Frag {
  union { v16bf v; uint4 q[2]; unsigned int u[8]; };
};

__device__ __forceinline__ unsigned short f2bf(float f) {
  unsigned u = __builtin_bit_cast(unsigned, f);
  u += 0x7FFFu + ((u >> 16) & 1u);          // round-to-nearest-even
  return (unsigned short)(u >> 16);
}
__device__ __forceinline__ unsigned pack2(float lo, float hi) {
  return (unsigned)f2bf(lo) | ((unsigned)f2bf(hi) << 16);
}

// LDS byte offset of a generic pointer to __shared__ (low 32 bits of flat addr)
__device__ __forceinline__ unsigned lds_off(const void* p) {
  return (unsigned)(unsigned long long)p;
}

// ---- CDNA5 async global->LDS copy (ASYNCcnt) -------------------------------
__device__ __forceinline__ void async_copy16(unsigned ldsaddr, const void* g) {
  asm volatile("global_load_async_to_lds_b128 %0, %1, off"
               :: "v"(ldsaddr), "v"(g) : "memory");
}
__device__ __forceinline__ void wait_async0() {
  asm volatile("s_wait_asynccnt 0x0" ::: "memory");
}

// ---- CDNA5 Tensor Data Mover: 2-D fp32 tile (rows x 768) -> LDS ------------
// D# per CDNA5 ISA. TENSORcnt-tracked.
// This toolchain's builtin takes 6 args (g0, g1, g2, g3, g4, cpol).
__device__ __forceinline__ void tdm_load_bias_tile(unsigned ldsaddr,
                                                   const float* gsrc) {
  unsigned long long ga = (unsigned long long)gsrc;
  v4u g0;
  g0[0] = 1u;                                        // count=1, user mode
  g0[1] = ldsaddr;                                   // lds_addr [63:32]
  g0[2] = (unsigned)(ga & 0xFFFFFFFFu);              // global_addr[31:0]
  g0[3] = (unsigned)((ga >> 32) & 0x01FFFFFFu)       // global_addr[56:32]
          | 0x80000000u;                             // type=2 ("image")
  v8i g1;
  g1[0] = (2 << 16);                                 // data_size=4B, no flags
  g1[1] = (int)((S_DIM & 0xFFFF) << 16);             // tensor_dim0 lo16
  g1[2] = (int)((T_DIM & 0xFFFF) << 16);             // dim0 hi=0 | tensor_dim1 lo16
  g1[3] = (int)(S_DIM << 16);                        // dim1 hi=0 | tile_dim0=768
  g1[4] = 16;                                        // tile_dim1=16, tile_dim2=0
  g1[5] = S_DIM;                                     // tensor_dim0_stride lo32
  g1[6] = 0;                                         // stride hi | dim1_stride lo
  g1[7] = 0;
  v4i z4; z4[0] = 0; z4[1] = 0; z4[2] = 0; z4[3] = 0;   // groups 2/3 unused (2-D)
  v8i z8;
  for (int i = 0; i < 8; i++) z8[i] = 0;
  __builtin_amdgcn_tensor_load_to_lds(g0, g1, z4, z4, z8, 0);
}

// ---------------------------------------------------------------- convert
__global__ void k_convert(const float* __restrict__ src,
                          unsigned short* __restrict__ dst, int n) {
  int i = blockIdx.x * 256 + threadIdx.x;
  if (i < n) dst[i] = f2bf(src[i]);
}

// ---------------------------------------------------------------- GEMM core
// C[128x128] tile of A[M,K] * B[N,K]^T, A/B bf16 row-major (K contiguous).
// 8 waves: wave tile 32(M) x 64(N) = 2x4 subtiles of 16x16.
// Staging is done with GLOBAL_LOAD_ASYNC_TO_LDS_B128 (4 per thread per step).
__device__ __forceinline__ void gemm_core(const unsigned short* __restrict__ A,
                                          const unsigned short* __restrict__ Bm,
                                          int mBlk, int nBlk, v8f acc[2][4]) {
  __shared__ __align__(16) unsigned short sA[128 * 32];
  __shared__ __align__(16) unsigned short sB[128 * 32];
  const int tid  = threadIdx.x;
  const int wave = tid >> 5, lane = tid & 31;
  const int g = lane >> 4, lm = lane & 15;
  const int wm = (wave >> 1) * 32;
  const int wn = (wave & 1) * 64;

  v8f zero; for (int r = 0; r < 8; r++) zero[r] = 0.f;
  for (int mi = 0; mi < 2; mi++)
    for (int ni = 0; ni < 4; ni++) acc[mi][ni] = zero;

  // chunk tid covers (row=tid/4, 16B slice tid%4); chunk tid+256 -> row+64
  const int r0 = tid >> 2, o0 = (tid & 3) * 8;   // elements
  const unsigned short* Ab0 = A  + (size_t)(mBlk + r0)      * E_DIM + o0;
  const unsigned short* Ab1 = A  + (size_t)(mBlk + r0 + 64) * E_DIM + o0;
  const unsigned short* Bb0 = Bm + (size_t)(nBlk + r0)      * E_DIM + o0;
  const unsigned short* Bb1 = Bm + (size_t)(nBlk + r0 + 64) * E_DIM + o0;
  const unsigned ldsA0 = lds_off(sA) + tid * 16;
  const unsigned ldsA1 = ldsA0 + 4096;
  const unsigned ldsB0 = lds_off(sB) + tid * 16;
  const unsigned ldsB1 = ldsB0 + 4096;

  const uint4* cA = (const uint4*)sA;
  const uint4* cB = (const uint4*)sB;

  for (int kb = 0; kb < E_DIM; kb += 32) {
    async_copy16(ldsA0, Ab0 + kb);
    async_copy16(ldsA1, Ab1 + kb);
    async_copy16(ldsB0, Bb0 + kb);
    async_copy16(ldsB1, Bb1 + kb);
    wait_async0();
    __syncthreads();

    Frag af[2], bf[4];
    for (int mi = 0; mi < 2; mi++) {
      int row = wm + mi * 16 + lm;                 // A: M per lane, K pairs
      af[mi].q[0] = cA[row * 4 + g];               // K = g*8 .. +7
      af[mi].q[1] = cA[row * 4 + g + 2];           // K = 16+g*8 .. +7
    }
    for (int ni = 0; ni < 4; ni++) {
      int row = wn + ni * 16 + lm;                 // B: N per lane, K halves
      bf[ni].q[0] = cB[row * 4 + g];               // K = g*8 .. +7
      bf[ni].q[1] = cB[row * 4 + g + 2];           // K = 16+g*8 .. +7
    }
    for (int mi = 0; mi < 2; mi++)
      for (int ni = 0; ni < 4; ni++)
        acc[mi][ni] = __builtin_amdgcn_wmma_f32_16x16x32_bf16(
            false, af[mi].v, false, bf[ni].v, (short)0, acc[mi][ni], false, false);
    __syncthreads();
  }
}

// ---------------------------------------------------------------- QKV GEMM
__global__ void __launch_bounds__(256) k_qkv(
    const unsigned short* __restrict__ qryB,
    const unsigned short* __restrict__ WqB,
    const unsigned short* __restrict__ WkB,
    const unsigned short* __restrict__ WvB,
    const float* __restrict__ bq, const float* __restrict__ bv,
    unsigned short* __restrict__ qB, unsigned short* __restrict__ kB,
    unsigned short* __restrict__ vT) {
  int mode = blockIdx.z;
  const unsigned short* Bm = (mode == 0) ? WqB : ((mode == 1) ? WkB : WvB);
  v8f acc[2][4];
  int mBlk = blockIdx.x * 128, nBlk = blockIdx.y * 128;
  gemm_core(qryB, Bm, mBlk, nBlk, acc);

  const int lane = threadIdx.x & 31, wave = threadIdx.x >> 5;
  const int g = lane >> 4, lm = lane & 15;
  const int wm = (wave >> 1) * 32, wn = (wave & 1) * 64;
  for (int mi = 0; mi < 2; mi++)
    for (int ni = 0; ni < 4; ni++)
      for (int r = 0; r < 8; r++) {
        int gm = mBlk + wm + mi * 16 + r + g * 8;   // row = t*B + b
        int ge = nBlk + wn + ni * 16 + lm;          // e_out
        int t = gm / B_DIM, b = gm % B_DIM;
        int h = ge >> 5, d = ge & 31;
        float val = acc[mi][ni][r];
        if (mode == 0) {
          val = (val + bq[ge]) * SCALING;
          qB[(((size_t)b * H_DIM + h) * T_DIM + t) * DH + d] = f2bf(val);
        } else if (mode == 1) {
          kB[(((size_t)b * H_DIM + h) * S_DIM + t) * DH + d] = f2bf(val);
        } else {
          val += bv[ge];
          vT[(((size_t)b * H_DIM + h) * DH + d) * S_DIM + t] = f2bf(val);
        }
      }
}

// ---------------------------------------------------------------- PBC expand
__global__ void k_expand(const int* __restrict__ outcell,
                         unsigned short* __restrict__ kB,
                         unsigned short* __restrict__ vT) {
  int tid = blockIdx.x * 256 + threadIdx.x;   // B*H*NEXP*16 threads
  int c   = tid & 15;
  int row = tid >> 4;
  int er  = row % NEXP_;
  int bh  = row / NEXP_;
  int b   = bh / H_DIM;
  int idx = outcell[b * NEXP_ + er] + 1;      // +1 offset per reference
  unsigned int* kU = (unsigned int*)kB;
  kU[((size_t)bh * S_DIM + T_DIM + er) * 16 + c] =
      kU[((size_t)bh * S_DIM + idx) * 16 + c];
  int d0 = c * 2;
  vT[((size_t)bh * DH + d0) * S_DIM + (T_DIM + er)] =
      vT[((size_t)bh * DH + d0) * S_DIM + idx];
  vT[((size_t)bh * DH + d0 + 1) * S_DIM + (T_DIM + er)] =
      vT[((size_t)bh * DH + d0 + 1) * S_DIM + idx];
}

// ---------------------------------------------------------------- attention
// grid (T/16, B), 256 threads (8 waves). Per head: scores -> softmax -> P*V.
// attn_bias tiles (the dominant 402MB HBM stream) are TDM-prefetched into a
// double-buffered LDS region one head ahead, overlapping DMA with compute.
__global__ void __launch_bounds__(256) k_attn(
    const unsigned short* __restrict__ qB,
    const unsigned short* __restrict__ kB,
    const unsigned short* __restrict__ vT,
    const float* __restrict__ bias,
    const unsigned char* __restrict__ kpm,
    const unsigned char* __restrict__ em,
    float* __restrict__ attn,    // [T,B,E]
    float* __restrict__ pmean) { // [B,T,S]
  extern __shared__ char smemRaw[];
  float* sc  = (float*)smemRaw;            // 16 x 768 scores/probs
  float* ps  = sc + 16 * S_DIM;            // 16 x 768 head-summed probs
  float* bb  = ps + 16 * S_DIM;            // 2 x (16 x 768) bias tiles (TDM)
  float* red = bb + 2 * 16 * S_DIM;        // 8 x 16 x 32 PV partials
  float* rr  = red + 8 * 512;              // 16 x 16 row reductions

  const int tBase = blockIdx.x * 16;
  const int b = blockIdx.y;
  const int tid = threadIdx.x;
  const int lane = tid & 31;
  const int g = lane >> 4, lm = lane & 15;
  // scalar (SGPR) wave id -> provably uniform branch for single-issue TDM
  const int wv = __builtin_amdgcn_readfirstlane(tid) >> 5;
  const int wave = tid >> 5;

  const float* biasBlk = bias + ((size_t)b * H_DIM) * T_DIM * S_DIM +
                         (size_t)tBase * S_DIM;

  for (int e = tid; e < 16 * S_DIM; e += 256) ps[e] = 0.f;

  if (wv == 0) {  // prefetch head 0 bias tile
    tdm_load_bias_tile(lds_off(bb), biasBlk);
  }
  __syncthreads();

  for (int h = 0; h < H_DIM; h++) {
    const unsigned short* qh = qB + (((size_t)b * H_DIM + h) * T_DIM) * DH;
    const unsigned short* kh = kB + (((size_t)b * H_DIM + h) * S_DIM) * DH;
    const unsigned short* vh = vT + (((size_t)b * H_DIM + h) * DH) * S_DIM;
    const float* bt = bb + (h & 1) * 16 * S_DIM;

    if (wv == 0) {
      if (h + 1 < H_DIM)   // issue next head's tile, then wait current ready
        tdm_load_bias_tile(lds_off(bb + ((h + 1) & 1) * 16 * S_DIM),
                           biasBlk + (size_t)(h + 1) * T_DIM * S_DIM);
      if (h == H_DIM - 1) __builtin_amdgcn_s_wait_tensorcnt(0);
      else                __builtin_amdgcn_s_wait_tensorcnt(1);
    }
    __syncthreads();   // bias tile for head h visible to all waves

    Frag aq;  // Q tile 16x32
    {
      const uint4* qg = (const uint4*)qh;
      int row = tBase + lm;
      aq.q[0] = qg[row * 4 + g];
      aq.q[1] = qg[row * 4 + g + 2];
    }
    const uint4* kg = (const uint4*)kh;
    for (int ct = 0; ct < 6; ct++) {           // wave's 6 score column tiles
      int sB0 = wave * 96 + ct * 16;
      Frag bk;
      int row = sB0 + lm;
      bk.q[0] = kg[row * 4 + g];
      bk.q[1] = kg[row * 4 + g + 2];
      v8f zero; for (int r = 0; r < 8; r++) zero[r] = 0.f;
      v8f s = __builtin_amdgcn_wmma_f32_16x16x32_bf16(
          false, aq.v, false, bk.v, (short)0, zero, false, false);
      for (int r = 0; r < 8; r++) {
        int m = r + g * 8;
        int col = sB0 + lm;
        float val = s[r] + bt[m * S_DIM + col];
        bool msk = (col < T_DIM) ? (kpm[b * T_DIM + col] != 0)
                                 : (em[b * NEXP_ + (col - T_DIM)] != 0);
        if (msk) val = NEG_BIG;
        sc[m * S_DIM + col] = val;
      }
    }
    __syncthreads();

    {  // softmax: thread (r = tid/16, p = tid%16) owns 48 columns of row r
      int r = tid >> 4, p = tid & 15;
      float* rowp = sc + r * S_DIM + p * 48;
      float mx = NEG_BIG;
      for (int i = 0; i < 48; i++) mx = fmaxf(mx, rowp[i]);
      rr[tid] = mx;
      __syncthreads();
      float rmax = NEG_BIG;
      for (int q2 = 0; q2 < 16; q2++) rmax = fmaxf(rmax, rr[r * 16 + q2]);
      __syncthreads();
      float sum = 0.f;
      for (int i = 0; i < 48; i++) {
        float e2 = __expf(rowp[i] - rmax);
        rowp[i] = e2; sum += e2;
      }
      rr[tid] = sum;
      __syncthreads();
      float rs = 0.f;
      for (int q2 = 0; q2 < 16; q2++) rs += rr[r * 16 + q2];
      float inv = 1.f / rs;
      float* pp = ps + r * S_DIM + p * 48;
      for (int i = 0; i < 48; i++) {
        float pr = rowp[i] * inv;
        rowp[i] = pr; pp[i] += pr;
      }
    }
    __syncthreads();

    {  // P*V: wave owns K-range [wave*96, +96), accumulates 16x32 partial
      v8f c0, c1;
      for (int r = 0; r < 8; r++) { c0[r] = 0.f; c1[r] = 0.f; }
      const uint4* vg = (const uint4*)vh;
      for (int kc = 0; kc < 3; kc++) {
        int sB0 = wave * 96 + kc * 32;
        Frag ap;
        for (int j = 0; j < 8; j++) {
          int k = g * 16 + j * 2;
          float f0 = sc[lm * S_DIM + sB0 + k];
          float f1 = sc[lm * S_DIM + sB0 + k + 1];
          ap.u[j] = pack2(f0, f1);
        }
        Frag b0, b1;
        int d0 = lm, d1 = 16 + lm;
        size_t i0 = ((size_t)d0 * S_DIM + sB0 + g * 16) >> 3;
        size_t i1 = ((size_t)d1 * S_DIM + sB0 + g * 16) >> 3;
        b0.q[0] = vg[i0];     b0.q[1] = vg[i0 + 1];
        b1.q[0] = vg[i1];     b1.q[1] = vg[i1 + 1];
        c0 = __builtin_amdgcn_wmma_f32_16x16x32_bf16(
            false, ap.v, false, b0.v, (short)0, c0, false, false);
        c1 = __builtin_amdgcn_wmma_f32_16x16x32_bf16(
            false, ap.v, false, b1.v, (short)0, c1, false, false);
      }
      for (int r = 0; r < 8; r++) {
        int m = r + g * 8;
        red[(wave * 16 + m) * 32 + lm]      = c0[r];
        red[(wave * 16 + m) * 32 + 16 + lm] = c1[r];
      }
    }
    __syncthreads();
    for (int e = tid; e < 512; e += 256) {   // reduce 8 wave partials
      float sum = 0.f;
      for (int w2 = 0; w2 < 8; w2++) sum += red[w2 * 512 + e];
      int m = e >> 5, d = e & 31;
      attn[((size_t)(tBase + m) * B_DIM + b) * E_DIM + h * DH + d] = sum;
    }
    __syncthreads();
  }

  const float invH = 1.0f / H_DIM;
  for (int e = tid; e < 16 * S_DIM; e += 256) {
    int m = e / S_DIM, col = e % S_DIM;
    pmean[((size_t)b * T_DIM + tBase + m) * S_DIM + col] = ps[e] * invH;
  }
}

// ---------------------------------------------------------------- layernorm
__global__ void __launch_bounds__(256) k_ln(const float* __restrict__ attn,
                                            const float* __restrict__ gw,
                                            const float* __restrict__ bw,
                                            unsigned short* __restrict__ out) {
  __shared__ float buf[256];
  int row = blockIdx.x;
  const float* x = attn + (size_t)row * E_DIM;
  int tid = threadIdx.x;
  float vals[4];
  float s = 0.f;
  for (int i = 0; i < 4; i++) { vals[i] = x[tid + i * 256]; s += vals[i]; }
  buf[tid] = s; __syncthreads();
  for (int off = 128; off > 0; off >>= 1) {
    if (tid < off) buf[tid] += buf[tid + off];
    __syncthreads();
  }
  float mu = buf[0] * (1.0f / E_DIM);
  __syncthreads();
  float v = 0.f;
  for (int i = 0; i < 4; i++) { float d = vals[i] - mu; v += d * d; }
  buf[tid] = v; __syncthreads();
  for (int off = 128; off > 0; off >>= 1) {
    if (tid < off) buf[tid] += buf[tid + off];
    __syncthreads();
  }
  float inv = rsqrtf(buf[0] * (1.0f / E_DIM) + LN_EPS_);
  for (int i = 0; i < 4; i++) {
    int e = tid + i * 256;
    float y = (vals[i] - mu) * inv * gw[e] + bw[e];
    out[(size_t)row * E_DIM + e] = f2bf(y);
  }
}

// ---------------------------------------------------------------- out proj
__global__ void __launch_bounds__(256) k_oproj(
    const unsigned short* __restrict__ aln,
    const unsigned short* __restrict__ WoB,
    const float* __restrict__ bo, float* __restrict__ out) {
  v8f acc[2][4];
  int mBlk = blockIdx.x * 128, nBlk = blockIdx.y * 128;
  gemm_core(aln, WoB, mBlk, nBlk, acc);
  const int lane = threadIdx.x & 31, wave = threadIdx.x >> 5;
  const int g = lane >> 4, lm = lane & 15;
  const int wm = (wave >> 1) * 32, wn = (wave & 1) * 64;
  for (int mi = 0; mi < 2; mi++)
    for (int ni = 0; ni < 4; ni++)
      for (int r = 0; r < 8; r++) {
        int gm = mBlk + wm + mi * 16 + r + g * 8;
        int ge = nBlk + wn + ni * 16 + lm;
        out[(size_t)gm * E_DIM + ge] = acc[mi][ni][r] + bo[ge];
      }
}

// ---------------------------------------------------------------- launch
extern "C" void kernel_launch(void* const* d_in, const int* in_sizes, int n_in,
                              void* d_out, int out_size, void* d_ws, size_t ws_size,
                              hipStream_t stream) {
  const float* query      = (const float*)d_in[0];
  const float* attn_bias  = (const float*)d_in[1];
  const int*   outcell    = (const int*)d_in[2];
  const unsigned char* em  = (const unsigned char*)d_in[3];
  const unsigned char* kpm = (const unsigned char*)d_in[4];
  const float* Wq = (const float*)d_in[5];
  const float* bq = (const float*)d_in[6];
  const float* Wk = (const float*)d_in[7];
  const float* Wv = (const float*)d_in[8];
  const float* bv = (const float*)d_in[9];
  const float* Wo = (const float*)d_in[10];
  const float* bo = (const float*)d_in[11];
  const float* lng = (const float*)d_in[12];
  const float* lnb = (const float*)d_in[13];

  const size_t MB = 1ull << 20;
  char* ws = (char*)d_ws;
  unsigned short* qB     = (unsigned short*)(ws);              //  8 MiB
  unsigned short* kB     = (unsigned short*)(ws + 8  * MB);    // 12 MiB
  unsigned short* vT     = (unsigned short*)(ws + 20 * MB);    // 12 MiB
  float*          attnf  = (float*)        (ws + 32 * MB);     // 16 MiB
  unsigned short* attnLN = (unsigned short*)(ws + 48 * MB);    //  8 MiB
  unsigned short* queryB = (unsigned short*)(ws + 56 * MB);    //  8 MiB
  unsigned short* WqB    = (unsigned short*)(ws + 64 * MB);    //  2 MiB
  unsigned short* WkB    = (unsigned short*)(ws + 66 * MB);
  unsigned short* WvB    = (unsigned short*)(ws + 68 * MB);
  unsigned short* WoB    = (unsigned short*)(ws + 70 * MB);    // ends 72 MiB

  float* outp  = (float*)d_out;                                 // [T,B,E]
  float* pmean = outp + (size_t)T_DIM * B_DIM * E_DIM;          // [B,T,S]

  const int nq = T_DIM * B_DIM * E_DIM;   // 4194304
  const int nw = E_DIM * E_DIM;           // 1048576
  k_convert<<<nq / 256, 256, 0, stream>>>(query, queryB, nq);
  k_convert<<<nw / 256, 256, 0, stream>>>(Wq, WqB, nw);
  k_convert<<<nw / 256, 256, 0, stream>>>(Wk, WkB, nw);
  k_convert<<<nw / 256, 256, 0, stream>>>(Wv, WvB, nw);
  k_convert<<<nw / 256, 256, 0, stream>>>(Wo, WoB, nw);

  k_qkv<<<dim3(32, 8, 3), 256, 0, stream>>>(queryB, WqB, WkB, WvB, bq, bv,
                                            qB, kB, vT);
  k_expand<<<4096, 256, 0, stream>>>(outcell, kB, vT);

  // sc + ps + 2x bias tiles + red + rr
  size_t smem = (size_t)(16 * S_DIM * 4 + 8 * 512 + 256) * sizeof(float);
  k_attn<<<dim3(T_DIM / 16, B_DIM), 256, smem, stream>>>(
      qB, kB, vT, attn_bias, kpm, em, attnf, pmean);

  k_ln<<<T_DIM * B_DIM, 256, 0, stream>>>(attnf, lng, lnb, attnLN);
  k_oproj<<<dim3(32, 8), 256, 0, stream>>>(attnLN, WoB, bo, outp);
}
